// TokenSpaceBlock_41300405518694
// MI455X (gfx1250) — matrix operand order repaired and checked by reading, not compiled, not run on hardware
//
#include <hip/hip_runtime.h>
#include <hip/hip_bf16.h>
#include <math.h>

// ---------------------------------------------------------------- constants
#define BB 4
#define NN 32768
#define CC 256
#define HH 8
#define DD 32
#define GG 32
#define BN (BB * NN)            // 131072 rows
#define NCH (NN / 256)          // 128 chunks of 256 points
static const float SCALE_ATT = 0.17677669529663687f;   // 32^-0.5
#define EPS_LN 1e-5f
#define EPS_NORM 1e-5f

typedef __bf16 bf16_t;
typedef __attribute__((ext_vector_type(16))) __bf16 bf16x16;
typedef __attribute__((ext_vector_type(8)))  float  f32x8;

__device__ __forceinline__ f32x8 wmma_bf16(bf16x16 a, bf16x16 b, f32x8 c) {
    return __builtin_amdgcn_wmma_f32_16x16x32_bf16(false, a, false, b,
                                                   (short)0, c, false, false);
}

__device__ __forceinline__ f32x8 f32x8_zero() {
    f32x8 z;
    #pragma unroll
    for (int i = 0; i < 8; ++i) z[i] = 0.0f;
    return z;
}

// Async global->LDS 16B copy (GVS mode: uniform 64-bit base + per-lane u32
// byte offset). LDS dest = per-lane LDS byte address. Tracked by ASYNCcnt.
__device__ __forceinline__ void async_g2l_b128(void* lds_ptr,
                                               const void* base,
                                               uint32_t byte_off) {
    uint32_t lds_addr = (uint32_t)(uintptr_t)lds_ptr;  // low 32b = LDS addr
    asm volatile("global_load_async_to_lds_b128 %0, %1, %2"
                 :: "v"(lds_addr), "v"(byte_off), "s"(base)
                 : "memory");
}
__device__ __forceinline__ void wait_async0() {
    asm volatile("s_wait_asynccnt 0x0" ::: "memory");
}

// A fragment (16x32 bf16) from row-major [M][K] LDS tile.
// lanes 0-15: M=lane, K = half*8+{0..7} and 16+half*8+{0..7}
// -> two contiguous 16B runs per lane: 2x ds_load_b128.
__device__ __forceinline__ bf16x16 load_a_frag(const bf16_t* s, int ld,
                                               int m0, int k0, int lane) {
    int half = (lane >> 4) & 1, m = m0 + (lane & 15);
    const bf16_t* p = s + m * ld + k0 + (half << 3);
    bf16x16 a;
    *(uint4*)&a       = *(const uint4*)p;
    *((uint4*)&a + 1) = *(const uint4*)(p + 16);
    return a;
}

// B fragment (32x16 bf16) from an [N][K]-stored (transposed) LDS tile.
// lane layout: N=lane&15, K = half*8+{0..7} and 16+half*8+{0..7}: two 16B runs.
__device__ __forceinline__ bf16x16 load_bt_frag(const bf16_t* s, int ld,
                                                int n0, int k0, int lane) {
    int half = (lane >> 4) & 1, n = n0 + (lane & 15);
    const bf16_t* p = s + n * ld + k0 + (half << 3);
    bf16x16 b;
    *(uint4*)&b       = *(const uint4*)p;
    *((uint4*)&b + 1) = *(const uint4*)(p + 16);
    return b;
}

// ---------------------------------------------------------------- K0: weight packing (transposed)
__global__ __launch_bounds__(256) void k0_pack(const float* __restrict__ Wx,
                                               const float* __restrict__ Wfx,
                                               const float* __restrict__ Wo,
                                               bf16_t* __restrict__ WpackT,
                                               bf16_t* __restrict__ WoT) {
    int i = blockIdx.x * 256 + threadIdx.x;
    if (i < 512 * 256) {                       // WpackT[n][k] = W*[k][n]
        int n = i >> 8, k = i & 255;
        float v = (n < 256) ? Wx[k * 256 + n] : Wfx[k * 256 + (n - 256)];
        WpackT[i] = (bf16_t)v;
    }
    if (i < 256 * 256) {                       // WoT[n][k] = Wo[k][n]
        int n = i >> 8, k = i & 255;
        WoT[i] = (bf16_t)Wo[k * 256 + n];
    }
}

// ---------------------------------------------------------------- zero accumulators
__global__ __launch_bounds__(256) void k_zero(float* __restrict__ st_acc,
                                              float* __restrict__ nrm_acc) {
    int i = blockIdx.x * 256 + threadIdx.x;
    if (i < BB * HH * GG * DD) st_acc[i] = 0.0f;
    if (i < BB * HH * GG)      nrm_acc[i] = 0.0f;
}

// ---------------------------------------------------------------- K1: LayerNorm rows
__global__ __launch_bounds__(256) void k1_ln(const float* __restrict__ fx,
                                             const float* __restrict__ g,
                                             const float* __restrict__ b,
                                             bf16_t* __restrict__ fxn,
                                             float* __restrict__ mr) {
    int row  = blockIdx.x * 8 + (threadIdx.x >> 5);
    int lane = threadIdx.x & 31;
    const float* x = fx + (size_t)row * CC;
    float v[8], s = 0.f, s2 = 0.f;
    #pragma unroll
    for (int i = 0; i < 8; ++i) {
        v[i] = x[lane * 8 + i];
        s += v[i]; s2 += v[i] * v[i];
    }
    #pragma unroll
    for (int off = 16; off >= 1; off >>= 1) {
        s  += __shfl_xor(s,  off, 32);
        s2 += __shfl_xor(s2, off, 32);
    }
    float mu = s * (1.0f / CC);
    float var = s2 * (1.0f / CC) - mu * mu;
    float rs = rsqrtf(var + EPS_LN);
    if (lane == 0) { mr[2 * row] = mu; mr[2 * row + 1] = rs; }
    bf16_t* y = fxn + (size_t)row * CC;
    #pragma unroll
    for (int i = 0; i < 8; ++i) {
        int c = lane * 8 + i;
        y[c] = (bf16_t)((v[i] - mu) * rs * g[c] + b[c]);
    }
}

// ---------------------------------------------------------------- shared GEMM mainloop
// C[128,128] tile of A[BN,256] x BT[Ntot,256]^T, bf16 in, f32 acc.
// Async global->LDS double buffering (no staging registers), 8 waves in
// 4(M)x2(N) grid, per-wave 2x4 16x16x32 WMMA tiles.
__device__ __forceinline__ void gemm128_mainloop(
        const bf16_t* __restrict__ A, const bf16_t* __restrict__ BT,
        int m0, int n0, int tid, int lane, int wave,
        bf16_t As[2][128][40], bf16_t Bs[2][128][40], f32x8 acc[2][4]) {
    int wm = (wave >> 1) * 32, wn = (wave & 1) * 64;

    auto issue_tile = [&](int buf, int kt) {
        #pragma unroll
        for (int i = 0; i < 2; ++i) {
            int c = tid + i * 256, r = c >> 2, cc = (c & 3) * 8;
            async_g2l_b128(&As[buf][r][cc], A,
                           (uint32_t)(((m0 + r) * 256 + kt + cc) * 2));
            async_g2l_b128(&Bs[buf][r][cc], BT,
                           (uint32_t)(((n0 + r) * 256 + kt + cc) * 2));
        }
    };

    issue_tile(0, 0);
    wait_async0();
    __syncthreads();
    #pragma unroll
    for (int it = 0; it < 8; ++it) {
        int cur = it & 1;
        if (it < 7) issue_tile(cur ^ 1, (it + 1) * 32);   // overlap with WMMA
        bf16x16 af[2];
        #pragma unroll
        for (int i = 0; i < 2; ++i)
            af[i] = load_a_frag(&As[cur][0][0], 40, wm + i * 16, 0, lane);
        #pragma unroll
        for (int j = 0; j < 4; ++j) {
            bf16x16 bf = load_bt_frag(&Bs[cur][0][0], 40, wn + j * 16, 0, lane);
            acc[0][j] = wmma_bf16(af[0], bf, acc[0][j]);
            acc[1][j] = wmma_bf16(af[1], bf, acc[1][j]);
        }
        if (it < 7) {
            wait_async0();
            __syncthreads();
        }
    }
}

// ---------------------------------------------------------------- K2: [BN,256]x[256,512]
__global__ __launch_bounds__(256) void k2_gemm(const bf16_t* __restrict__ A,
                                               const bf16_t* __restrict__ BT,
                                               const float* __restrict__ bx,
                                               const float* __restrict__ bfx,
                                               bf16_t* __restrict__ Cout) {
    __shared__ bf16_t As[2][128][40];
    __shared__ bf16_t Bs[2][128][40];
    int tid = threadIdx.x, lane = tid & 31, wave = tid >> 5;
    int m0 = blockIdx.x * 128, n0 = blockIdx.y * 128;
    int wm = (wave >> 1) * 32, wn = (wave & 1) * 64;

    f32x8 acc[2][4];
    #pragma unroll
    for (int i = 0; i < 2; ++i)
        #pragma unroll
        for (int j = 0; j < 4; ++j) acc[i][j] = f32x8_zero();

    gemm128_mainloop(A, BT, m0, n0, tid, lane, wave, As, Bs, acc);

    int half = lane >> 4, nn = lane & 15;
    #pragma unroll
    for (int i = 0; i < 2; ++i)
        #pragma unroll
        for (int j = 0; j < 4; ++j) {
            int gn = n0 + wn + j * 16 + nn;
            float bias = (gn < 256) ? bx[gn] : bfx[gn - 256];
            #pragma unroll
            for (int r = 0; r < 8; ++r) {
                int gm = m0 + wm + i * 16 + half * 8 + r;
                Cout[(size_t)gm * 512 + gn] = (bf16_t)(acc[i][j][r] + bias);
            }
        }
}

// ---------------------------------------------------------------- K3: slice softmax + split-K token reduce
__global__ __launch_bounds__(256) void k3_slice(const bf16_t* __restrict__ xfx,
                                                const float* __restrict__ Ws,
                                                const float* __restrict__ bs,
                                                const float* __restrict__ temp,
                                                bf16_t* __restrict__ slice_w,
                                                float* __restrict__ st_acc,
                                                float* __restrict__ nrm_acc) {
    __shared__ float  WsS[32][32];
    __shared__ float  bsS[32];
    __shared__ bf16_t WlT[32][264];   // [g][n_local] : A (transposed store)
    __shared__ bf16_t FlT[32][264];   // [d][n_local] : B in [N][K] layout
    int tid = threadIdx.x, lane = tid & 31, wave = tid >> 5;
    int nc = blockIdx.x & (NCH - 1);
    int bh = blockIdx.x >> 7;         // NCH == 128
    int h = bh & 7, b = bh >> 3;
    int n = nc * 256 + tid;

    #pragma unroll
    for (int i = 0; i < 4; ++i) {
        int e = tid + i * 256;
        (&WsS[0][0])[e] = Ws[e];
    }
    if (tid < 32) bsS[tid] = bs[tid];
    __syncthreads();

    const bf16_t* xrow = xfx + (size_t)(b * NN + n) * 512 + h * 32;
    float xm[32];
    #pragma unroll
    for (int d = 0; d < 32; ++d) xm[d] = (float)xrow[d];
    float tinv = 1.0f / temp[h];

    float lg[32], mx = -1e30f;
    #pragma unroll 1
    for (int g = 0; g < 32; ++g) {
        float s = bsS[g];
        #pragma unroll
        for (int d = 0; d < 32; ++d) s += xm[d] * WsS[d][g];
        s *= tinv;
        lg[g] = s;
        mx = fmaxf(mx, s);
    }
    float se = 0.f;
    #pragma unroll
    for (int g = 0; g < 32; ++g) { lg[g] = __expf(lg[g] - mx); se += lg[g]; }
    float inv = 1.0f / se;
    bf16_t* wrow = slice_w + ((size_t)bh * NN + n) * 32;
    #pragma unroll
    for (int g = 0; g < 32; ++g) {
        bf16_t wv = (bf16_t)(lg[g] * inv);
        wrow[g] = wv;
        WlT[g][tid] = wv;
    }
    const bf16_t* frow = xfx + (size_t)(b * NN + n) * 512 + 256 + h * 32;
    #pragma unroll
    for (int d = 0; d < 32; ++d) FlT[d][tid] = frow[d];
    __syncthreads();

    // slice_norm partial: row sums of WlT (contiguous)
    if (tid < 32) {
        float s = 0.f;
        #pragma unroll 1
        for (int i = 0; i < 256; ++i) s += (float)WlT[tid][i];
        atomicAdd(&nrm_acc[bh * 32 + tid], s);
    }

    // split-K WMMA: out[g][d] += sum_n WlT[g][n] * FlT[d][n]  (K = 256)
    if (wave < 4) {
        int gi = (wave >> 1) * 16, di = (wave & 1) * 16;
        f32x8 acc = f32x8_zero();
        #pragma unroll
        for (int ks = 0; ks < 8; ++ks) {
            bf16x16 a  = load_a_frag (&WlT[0][0], 264, gi, ks * 32, lane);
            bf16x16 bb = load_bt_frag(&FlT[0][0], 264, di, ks * 32, lane);
            acc = wmma_bf16(a, bb, acc);
        }
        int half = lane >> 4, nnn = lane & 15;
        #pragma unroll
        for (int r = 0; r < 8; ++r) {
            int g = gi + half * 8 + r, d = di + nnn;
            atomicAdd(&st_acc[((size_t)bh * 32 + g) * 32 + d], acc[r]);
        }
    }
}

// ---------------------------------------------------------------- K4: tiny attention + MLP over G tokens
__global__ __launch_bounds__(32) void k4_tok(const float* __restrict__ st_acc,
                                             const float* __restrict__ nrm_acc,
                                             const float* __restrict__ Wq,
                                             const float* __restrict__ Wk,
                                             const float* __restrict__ Wv,
                                             const float* __restrict__ tg,
                                             const float* __restrict__ tb,
                                             const float* __restrict__ W1,
                                             const float* __restrict__ b1,
                                             const float* __restrict__ W2,
                                             const float* __restrict__ b2,
                                             bf16_t* __restrict__ tokensT) {
    int bh = blockIdx.x, g = threadIdx.x;
    __shared__ float ksh[32][33], vsh[32][33];
    float st[32];
    float nrm = 1.0f / (nrm_acc[bh * 32 + g] + EPS_NORM);
    #pragma unroll
    for (int d = 0; d < 32; ++d)
        st[d] = st_acc[((size_t)bh * 32 + g) * 32 + d] * nrm;

    float q[32];
    #pragma unroll 1
    for (int d = 0; d < 32; ++d) {
        float sq = 0.f, sk = 0.f, sv = 0.f;
        #pragma unroll
        for (int e = 0; e < 32; ++e) {
            float se = st[e];
            sq += se * Wq[e * 32 + d];
            sk += se * Wk[e * 32 + d];
            sv += se * Wv[e * 32 + d];
        }
        q[d] = sq; ksh[g][d] = sk; vsh[g][d] = sv;
    }
    __syncthreads();

    float a[32], mx = -1e30f;
    #pragma unroll 1
    for (int j = 0; j < 32; ++j) {
        float s = 0.f;
        #pragma unroll
        for (int d = 0; d < 32; ++d) s += q[d] * ksh[j][d];
        s *= SCALE_ATT;
        a[j] = s; mx = fmaxf(mx, s);
    }
    float se = 0.f;
    #pragma unroll
    for (int j = 0; j < 32; ++j) { a[j] = __expf(a[j] - mx); se += a[j]; }
    float inv = 1.0f / se;

    float ot[32];
    #pragma unroll 1
    for (int d = 0; d < 32; ++d) {
        float s = 0.f;
        #pragma unroll
        for (int j = 0; j < 32; ++j) s += a[j] * vsh[j][d];
        ot[d] = s * inv + st[d];
    }

    // token LayerNorm over D
    float mu = 0.f, s2 = 0.f;
    #pragma unroll
    for (int d = 0; d < 32; ++d) { mu += ot[d]; s2 += ot[d] * ot[d]; }
    mu *= (1.0f / 32.0f);
    float rs = rsqrtf(s2 * (1.0f / 32.0f) - mu * mu + EPS_LN);
    float hn[32];
    #pragma unroll
    for (int d = 0; d < 32; ++d) hn[d] = (ot[d] - mu) * rs * tg[d] + tb[d];

    float tok[32];
    #pragma unroll
    for (int d = 0; d < 32; ++d) tok[d] = ot[d] + b2[d];
    #pragma unroll 1
    for (int x = 0; x < 128; ++x) {
        float t = b1[x];
        #pragma unroll
        for (int d = 0; d < 32; ++d) t += hn[d] * W1[d * 128 + x];
        float ge = 0.5f * t * (1.0f + erff(t * 0.70710678118f));
        #pragma unroll
        for (int d = 0; d < 32; ++d) tok[d] += ge * W2[x * 32 + d];
    }
    // write transposed: tokensT[bh][d][g] (lanes over g -> coalesced)
    #pragma unroll
    for (int d = 0; d < 32; ++d)
        tokensT[((size_t)bh * 32 + d) * 32 + g] = (bf16_t)tok[d];
}

// ---------------------------------------------------------------- K5: combine tokens back to points
__global__ __launch_bounds__(256) void k5_comb(const bf16_t* __restrict__ slice_w,
                                               const bf16_t* __restrict__ tokensT,
                                               bf16_t* __restrict__ out_x) {
    __shared__ bf16_t Wl[256][40];   // [n_local][g]  : A row-major
    __shared__ bf16_t TlT[32][40];   // [d][g]        : B in [N][K] layout
    int tid = threadIdx.x, lane = tid & 31, wave = tid >> 5;
    int nc = blockIdx.x & (NCH - 1);
    int bh = blockIdx.x >> 7;
    int h = bh & 7, b = bh >> 3;
    int n = nc * 256 + tid;

    // async-stage both tiles straight into LDS
    uint32_t swoff = (uint32_t)((((size_t)bh * NN + n) * 32) * 2);
    #pragma unroll
    for (int c = 0; c < 4; ++c)
        async_g2l_b128(&Wl[tid][c * 8], slice_w, swoff + c * 16);
    if (tid < 128) {
        int r = tid >> 2, cc = (tid & 3) * 8;
        async_g2l_b128(&TlT[r][cc], tokensT,
                       (uint32_t)((((size_t)bh * 32 + r) * 32 + cc) * 2));
    }
    wait_async0();
    __syncthreads();

    int m0 = wave * 32;
    bf16x16 af[2], bfg[2];
    #pragma unroll
    for (int i = 0; i < 2; ++i)
        af[i] = load_a_frag(&Wl[0][0], 40, m0 + i * 16, 0, lane);
    #pragma unroll
    for (int j = 0; j < 2; ++j)
        bfg[j] = load_bt_frag(&TlT[0][0], 40, j * 16, 0, lane);

    int half = lane >> 4, nnn = lane & 15;
    #pragma unroll
    for (int i = 0; i < 2; ++i)
        #pragma unroll
        for (int j = 0; j < 2; ++j) {
            f32x8 acc = wmma_bf16(af[i], bfg[j], f32x8_zero());
            #pragma unroll
            for (int r = 0; r < 8; ++r) {
                int gm = b * NN + nc * 256 + m0 + i * 16 + half * 8 + r;
                int gn = h * 32 + j * 16 + nnn;
                out_x[(size_t)gm * 256 + gn] = (bf16_t)acc[r];
            }
        }
}

// ---------------------------------------------------------------- K6: output projection + residual
__global__ __launch_bounds__(256) void k6_gemm(const bf16_t* __restrict__ A,
                                               const bf16_t* __restrict__ BT,
                                               const float* __restrict__ fx,
                                               const float* __restrict__ mr,
                                               const float* __restrict__ lng,
                                               const float* __restrict__ lnb,
                                               const float* __restrict__ bo,
                                               float* __restrict__ out) {
    __shared__ bf16_t As[2][128][40];
    __shared__ bf16_t Bs[2][128][40];
    int tid = threadIdx.x, lane = tid & 31, wave = tid >> 5;
    int m0 = blockIdx.x * 128, n0 = blockIdx.y * 128;
    int wm = (wave >> 1) * 32, wn = (wave & 1) * 64;

    f32x8 acc[2][4];
    #pragma unroll
    for (int i = 0; i < 2; ++i)
        #pragma unroll
        for (int j = 0; j < 4; ++j) acc[i][j] = f32x8_zero();

    gemm128_mainloop(A, BT, m0, n0, tid, lane, wave, As, Bs, acc);

    int half = lane >> 4, nn = lane & 15;
    #pragma unroll
    for (int i = 0; i < 2; ++i)
        #pragma unroll
        for (int j = 0; j < 4; ++j) {
            int gn = n0 + wn + j * 16 + nn;
            float bov = bo[gn], gv = lng[gn], bv = lnb[gn];
            #pragma unroll
            for (int r = 0; r < 8; ++r) {
                int gm = m0 + wm + i * 16 + half * 8 + r;
                float mu = mr[2 * gm], rs = mr[2 * gm + 1];
                float fxv = fx[(size_t)gm * 256 + gn];
                out[(size_t)gm * 256 + gn] =
                    acc[i][j][r] + bov + (fxv - mu) * rs * gv + bv;
            }
        }
}

// ---------------------------------------------------------------- host
extern "C" void kernel_launch(void* const* d_in, const int* in_sizes, int n_in,
                              void* d_out, int out_size, void* d_ws, size_t ws_size,
                              hipStream_t stream) {
    const float* fx   = (const float*)d_in[0];
    const float* ln_g = (const float*)d_in[1];
    const float* ln_b = (const float*)d_in[2];
    const float* Wx   = (const float*)d_in[3];
    const float* bx   = (const float*)d_in[4];
    const float* Wfx  = (const float*)d_in[5];
    const float* bfx  = (const float*)d_in[6];
    const float* Ws   = (const float*)d_in[7];
    const float* bs   = (const float*)d_in[8];
    const float* temp = (const float*)d_in[9];
    const float* Wq   = (const float*)d_in[10];
    const float* Wk   = (const float*)d_in[11];
    const float* Wv   = (const float*)d_in[12];
    const float* tlg  = (const float*)d_in[13];
    const float* tlb  = (const float*)d_in[14];
    const float* W1   = (const float*)d_in[15];
    const float* b1   = (const float*)d_in[16];
    const float* W2   = (const float*)d_in[17];
    const float* b2   = (const float*)d_in[18];
    const float* Wo   = (const float*)d_in[19];
    const float* bo   = (const float*)d_in[20];
    float* out = (float*)d_out;

    // workspace carve-out (256B aligned)
    char* ws = (char*)d_ws;
    size_t off = 0;
    auto take = [&](size_t bytes) {
        char* p = ws + off;
        off = (off + bytes + 255) & ~(size_t)255;
        return p;
    };
    bf16_t* fxn    = (bf16_t*)take((size_t)BN * CC * 2);           // 64 MB
    float*  mr     = (float*) take((size_t)BN * 2 * 4);            // 1 MB
    bf16_t* WpackT = (bf16_t*)take((size_t)512 * 256 * 2);
    bf16_t* WoT    = (bf16_t*)take((size_t)256 * 256 * 2);
    bf16_t* xfx    = (bf16_t*)take((size_t)BN * 512 * 2);          // 128 MB
    bf16_t* sw     = (bf16_t*)take((size_t)BB * HH * NN * GG * 2); // 64 MB
    float*  st_acc = (float*) take((size_t)BB * HH * GG * DD * 4);
    float*  nrmac  = (float*) take((size_t)BB * HH * GG * 4);
    bf16_t* tokT   = (bf16_t*)take((size_t)BB * HH * GG * DD * 2);
    bf16_t* out_x  = (bf16_t*)take((size_t)BN * 256 * 2);          // 64 MB

    k0_pack<<<512, 256, 0, stream>>>(Wx, Wfx, Wo, WpackT, WoT);
    k_zero<<<128, 256, 0, stream>>>(st_acc, nrmac);
    k1_ln<<<BN / 8, 256, 0, stream>>>(fx, ln_g, ln_b, fxn, mr);
    k2_gemm<<<dim3(BN / 128, 4), 256, 0, stream>>>(fxn, WpackT, bx, bfx, xfx);
    k3_slice<<<BB * HH * NCH, 256, 0, stream>>>(xfx, Ws, bs, temp, sw,
                                                st_acc, nrmac);
    k4_tok<<<BB * HH, 32, 0, stream>>>(st_acc, nrmac, Wq, Wk, Wv, tlg, tlb,
                                       W1, b1, W2, b2, tokT);
    k5_comb<<<BB * HH * NCH, 256, 0, stream>>>(sw, tokT, out_x);
    k6_gemm<<<dim3(BN / 128, 2), 256, 0, stream>>>(out_x, WoT, fx, mr,
                                                   ln_g, ln_b, bo, out);
    (void)in_sizes; (void)n_in; (void)out_size; (void)ws_size;
}
